// SlotConvEncoder_90271622627394
// MI455X (gfx1250) — matrix operand order repaired and checked by reading, not compiled, not run on hardware
//
#include <hip/hip_runtime.h>
#include <hip/hip_bf16.h>
#include <math.h>

// Problem constants (from reference)
#define BS_TOTAL 16384   // B*S
#define GG 8             // groups (conv spatial axis)
#define DD 128           // d_chunk
#define HH 256           // conv hidden channels
#define DMM 1024         // d_model
#define NS 4             // samples per workgroup in encoder kernel

typedef __attribute__((ext_vector_type(16))) __bf16 v16bf;
typedef __attribute__((ext_vector_type(8)))  __bf16 v8bf;
typedef __attribute__((ext_vector_type(8)))  float  v8f;

// ---------------------------------------------------------------------------
// WMMA helpers (CDNA5 16x16x32 bf16, f32 accumulate)
// ---------------------------------------------------------------------------
__device__ __forceinline__ v8f wmma_bf16(v16bf a, v16bf b, v8f c) {
  // (neg_a, A, neg_b, B, c_mod, C, reuse_a, reuse_b)
  return __builtin_amdgcn_wmma_f32_16x16x32_bf16(false, a, false, b, (short)0, c,
                                                 false, false);
}

// A fragment (16x32 bf16). lane l: m = l&15, half = l>>4.
// element j -> K = kbase + 16*half + j
// => two contiguous 16B loads per lane.
__device__ __forceinline__ v16bf frag_a(const __bf16* row, int kbase, int half) {
  union { v16bf v; v8bf p[2]; } u;
  u.p[0] = *(const v8bf*)(row + kbase + 16 * half);
  u.p[1] = *(const v8bf*)(row + kbase + 16 * half + 8);
  return u.v;
}

// B fragment (32x16 bf16) from swizzled weights laid out [kcIdx][N][32]:
// lane l: n = l&15 (+tile base), half = l>>4; element j -> K = kc + 16*half + j
// => 32 contiguous bytes per lane.
__device__ __forceinline__ v16bf frag_b(const __bf16* wswz, int N, int kc, int n,
                                        int half) {
  const __bf16* p = wswz + ((size_t)(kc >> 5) * N + n) * 32 + half * 16;
  union { v16bf v; v8bf p2[2]; } u;
  u.p2[0] = *(const v8bf*)(p);
  u.p2[1] = *(const v8bf*)(p + 8);
  return u.v;
}

__device__ __forceinline__ float wred(float v) {
#pragma unroll
  for (int mk = 16; mk > 0; mk >>= 1) v += __shfl_xor(v, mk, 32);
  return v;
}

// ---------------------------------------------------------------------------
// CDNA5 async DMA: global -> LDS, tracked by ASYNCcnt (ISA 15.18.3 op 98).
// lds_addr: allocation-relative LDS byte offset in a VGPR (low 32 bits of a
// generic pointer to __shared__); voff: per-lane byte offset; sbase: SGPR pair.
// ---------------------------------------------------------------------------
__device__ __forceinline__ void async_ld_b128(unsigned lds_addr, unsigned voff,
                                              const void* sbase) {
  asm volatile("global_load_async_to_lds_b128 %0, %1, %2"
               :
               : "v"(lds_addr), "v"(voff), "s"(sbase)
               : "memory");
}

// ---------------------------------------------------------------------------
// Weight prep: fp32 -> bf16, transposed per tap, swizzled to [kcIdx][N][32]
// ---------------------------------------------------------------------------
__global__ __launch_bounds__(256) void swz_w1(const float* __restrict__ w1,
                                              __bf16* __restrict__ dst) {
  int i = blockIdx.x * 256 + threadIdx.x;          // 3 * 128 * 256
  if (i >= 3 * DD * HH) return;
  int tap = i / (DD * HH);
  int o   = i - tap * (DD * HH);
  int pos = o & 31;
  int n   = (o >> 5) & (HH - 1);                    // oc
  int kci = o >> 13;                                // / (32*256)
  int k   = kci * 32 + pos;                         // ic
  dst[i] = (__bf16)w1[(n * DD + k) * 3 + tap];      // w1[oc][ic][tap]
}

__global__ __launch_bounds__(256) void swz_w2(const float* __restrict__ w2,
                                              __bf16* __restrict__ dst) {
  int i = blockIdx.x * 256 + threadIdx.x;          // 3 * 256 * 128
  if (i >= 3 * HH * DD) return;
  int tap = i / (HH * DD);
  int o   = i - tap * (HH * DD);
  int pos = o & 31;
  int n   = (o >> 5) & (DD - 1);                    // od
  int kci = o >> 12;                                // / (32*128)
  int k   = kci * 32 + pos;                         // ih
  dst[i] = (__bf16)w2[(n * HH + k) * 3 + tap];      // w2[od][ih][tap]
}

__global__ __launch_bounds__(256) void swz_wm(const float* __restrict__ Wm,
                                              __bf16* __restrict__ dst) {
  int i = blockIdx.x * 256 + threadIdx.x;          // 1024 * 1024
  if (i >= DMM * DMM) return;
  int pos = i & 31;
  int n   = (i >> 5) & (DMM - 1);
  int kci = i >> 15;                                // / (32*1024)
  int k   = kci * 32 + pos;
  dst[i] = (__bf16)Wm[k * DMM + n];                 // Wm[p][m] row-major [K][N]
}

// ---------------------------------------------------------------------------
// Fused: gate/bias -> LN(D) -> conv1(+GELU) -> conv2(+residual) -> packLN
// One workgroup = NS=4 (b,s) samples = 32 GEMM rows. 8 waves of 32.
// Convs = shifted WMMA GEMMs over padded LDS buffers.
// ---------------------------------------------------------------------------
__global__ __launch_bounds__(256) void slotconv_encoder(
    const float* __restrict__ x, const float* __restrict__ slot_gate,
    const float* __restrict__ slot_bias, const float* __restrict__ pre_g,
    const float* __restrict__ pre_b, const __bf16* __restrict__ w1b,
    const float* __restrict__ b1, const __bf16* __restrict__ w2b,
    const float* __restrict__ b2, const float* __restrict__ pack_g,
    const float* __restrict__ pack_b, __bf16* __restrict__ hpk) {
  __shared__ __align__(16) float  resid[NS][GG * DD];   // 16 KB  gated input
  __shared__ __align__(16) __bf16 xln[NS][GG + 2][DD];  // 10 KB  LN1, zero-padded
  __shared__ __align__(16) __bf16 h1p[NS][GG + 2][HH];  // 20 KB  gelu(conv1), padded
  __shared__ __align__(16) float  ypk[NS][GG * DD];     // 16 KB  conv2+residual

  const int tid = threadIdx.x;
  const int lane = tid & 31;
  const int wave = tid >> 5;
  const int half = lane >> 4;
  const int m = lane & 15;
  const int rowBase = blockIdx.x * NS;

  // ---- gate + bias (residual kept in fp32) ----
  for (int t = tid; t < NS * GG * DD; t += 256) {
    int s = t >> 10;
    int p = t & 1023;
    float gate = 2.0f / (1.0f + __expf(-slot_gate[p]));
    resid[s][p] = x[(rowBase + s) * 1024 + p] * gate + slot_bias[p];
  }
  // zero conv padding rows (g = -1 and g = 8)
  for (int t = tid; t < NS * 2 * DD; t += 256) {
    int s = t >> 8, r = (t >> 7) & 1, d = t & 127;
    xln[s][r * (GG + 1)][d] = (__bf16)0.0f;
  }
  for (int t = tid; t < NS * 2 * HH; t += 256) {
    int s = t >> 9, r = (t >> 8) & 1, d = t & 255;
    h1p[s][r * (GG + 1)][d] = (__bf16)0.0f;
  }
  __syncthreads();

  // ---- LayerNorm over D per (s,g): one wave32 per 128-vector ----
#pragma unroll
  for (int it = 0; it < 4; ++it) {
    int vec = wave * 4 + it;  // 0..31
    int s = vec >> 3, g = vec & 7;
    float v[4], sum = 0.f, sq = 0.f;
#pragma unroll
    for (int j = 0; j < 4; ++j) {
      v[j] = resid[s][g * DD + lane + 32 * j];
      sum += v[j];
      sq += v[j] * v[j];
    }
    sum = wred(sum);
    sq = wred(sq);
    float mean = sum * (1.f / 128.f);
    float rstd = rsqrtf(sq * (1.f / 128.f) - mean * mean + 1e-5f);
#pragma unroll
    for (int j = 0; j < 4; ++j) {
      int d = lane + 32 * j;
      xln[s][g + 1][d] = (__bf16)((v[j] - mean) * rstd * pre_g[d] + pre_b[d]);
    }
  }
  __syncthreads();

  // ---- conv1 (128->256, K=3) as 3 shifted WMMA GEMMs, + bias + exact GELU ----
  for (int task = wave; task < 32; task += 8) {  // 2 M-tiles x 16 N-tiles
    int mt = task >> 4, nt = task & 15;
    int row = mt * 16 + m, s = row >> 3, g = row & 7;
    v8f acc = {};
#pragma unroll
    for (int tap = 0; tap < 3; ++tap) {
      const __bf16* aRow = &xln[s][g + tap][0];
      const __bf16* wb = w1b + tap * (DD * HH);
#pragma unroll
      for (int kc = 0; kc < DD; kc += 32) {
        v16bf a = frag_a(aRow, kc, half);
        v16bf b = frag_b(wb, HH, kc, nt * 16 + m, half);
        acc = wmma_bf16(a, b, acc);
      }
    }
    int col = nt * 16 + m;
    float bias = b1[col];
#pragma unroll
    for (int r = 0; r < 8; ++r) {
      int orow = mt * 16 + r + 8 * half;
      int os = orow >> 3, og = orow & 7;
      float vg = acc[r] + bias;
      vg = vg * 0.5f * (1.0f + erff(vg * 0.70710678118f));  // exact GELU
      h1p[os][og + 1][col] = (__bf16)vg;
    }
  }
  __syncthreads();

  // ---- conv2 (256->128, K=3) + bias + residual ----
  for (int task = wave; task < 16; task += 8) {  // 2 M-tiles x 8 N-tiles
    int mt = task >> 3, nt = task & 7;
    int row = mt * 16 + m, s = row >> 3, g = row & 7;
    v8f acc = {};
#pragma unroll
    for (int tap = 0; tap < 3; ++tap) {
      const __bf16* aRow = &h1p[s][g + tap][0];
      const __bf16* wb = w2b + tap * (HH * DD);
#pragma unroll
      for (int kc = 0; kc < HH; kc += 32) {
        v16bf a = frag_a(aRow, kc, half);
        v16bf b = frag_b(wb, DD, kc, nt * 16 + m, half);
        acc = wmma_bf16(a, b, acc);
      }
    }
    int col = nt * 16 + m;
    float bias = b2[col];
#pragma unroll
    for (int r = 0; r < 8; ++r) {
      int orow = mt * 16 + r + 8 * half;
      int os = orow >> 3, og = orow & 7;
      ypk[os][og * DD + col] = acc[r] + bias + resid[os][og * DD + col];
    }
  }
  __syncthreads();

  // ---- packed LayerNorm over 1024, emit bf16 GEMM activations ----
  if (wave < NS) {
    int s = wave;
    float sum = 0.f, sq = 0.f;
    for (int j = 0; j < 32; ++j) {
      float v = ypk[s][lane + 32 * j];
      sum += v;
      sq += v * v;
    }
    sum = wred(sum);
    sq = wred(sq);
    float mean = sum * (1.f / 1024.f);
    float rstd = rsqrtf(sq * (1.f / 1024.f) - mean * mean + 1e-5f);
    size_t grow = (size_t)(rowBase + s) * 1024;
    for (int j = 0; j < 32; ++j) {
      int p = lane + 32 * j;
      hpk[grow + p] = (__bf16)((ypk[s][p] - mean) * rstd * pack_g[p] + pack_b[p]);
    }
  }
}

// ---------------------------------------------------------------------------
// out[16384,1024] = hpk[16384,1024] x Wm[1024,1024] + bm
// WG tile 64x128, 8 waves; each wave owns a 32x32 block (4 accumulators) for
// a 1:1:1 wmma:ds:vmem issue ratio. A chunks (64x64 bf16) are DMA'd to a
// double-buffered LDS tile with GLOBAL_LOAD_ASYNC_TO_LDS_B128 (ASYNCcnt),
// overlapping HBM latency with the WMMA chain. B streams from the
// L2-resident swizzled bf16 Wm (2 MB).
// ---------------------------------------------------------------------------
__global__ __launch_bounds__(256) void slotconv_gemm(
    const __bf16* __restrict__ hpk, const __bf16* __restrict__ wmb,
    const float* __restrict__ bm, float* __restrict__ out) {
  __shared__ __align__(16) __bf16 aT[2][64][64];  // 16 KB double buffer

  const int tid = threadIdx.x;
  const int lane = tid & 31;
  const int wave = tid >> 5;
  const int half = lane >> 4;
  const int m = lane & 15;
  const int mBase = blockIdx.y * 64;
  const int nBase = blockIdx.x * 128;

  const int mt0 = (wave >> 2) * 2;                   // {0,2}: two M-tiles/wave
  const int n0 = nBase + ((wave & 3) * 2) * 16 + m;  // two N-tiles/wave
  const int n1 = n0 + 16;

  v8f acc00 = {}, acc01 = {}, acc10 = {}, acc11 = {};

  // staging: 64x64 bf16 chunk = 8 KB; 32 B per thread = 2 async b128
  const int lr = tid >> 2;            // 0..63 row
  const int lc = (tid & 3) * 16;      // 0/16/32/48 col
  const unsigned voff0 = (unsigned)(lr * 2048 + lc * 2);  // bytes
  const void* sbase = (const void*)(hpk + (size_t)mBase * 1024);
  const unsigned lds0 = (unsigned)(size_t)&aT[0][lr][lc];
  const unsigned lds1 = (unsigned)(size_t)&aT[1][lr][lc];

  // prologue: DMA chunks 0 and 1 (2 async instr / wave / chunk)
  async_ld_b128(lds0, voff0, sbase);
  async_ld_b128(lds0 + 16, voff0 + 16, sbase);
  async_ld_b128(lds1, voff0 + 128, sbase);
  async_ld_b128(lds1 + 16, voff0 + 144, sbase);

#pragma unroll
  for (int ch = 0; ch < 16; ++ch) {
    // async completes in-order: <=2 outstanding leaves only chunk ch+1 in
    // flight, guaranteeing chunk ch has landed in LDS.
    if (ch < 15)
      asm volatile("s_wait_asynccnt 2" ::: "memory");
    else
      asm volatile("s_wait_asynccnt 0" ::: "memory");
    __syncthreads();

    const __bf16* buf = &aT[ch & 1][0][0];
    const __bf16* r0 = buf + (mt0 * 16 + m) * 64;
    const __bf16* r1 = r0 + 16 * 64;
#pragma unroll
    for (int kk = 0; kk < 2; ++kk) {
      int kc = ch * 64 + kk * 32;
      v16bf a0 = frag_a(r0, kk * 32, half);
      v16bf a1 = frag_a(r1, kk * 32, half);
      v16bf b0 = frag_b(wmb, DMM, kc, n0, half);
      v16bf b1 = frag_b(wmb, DMM, kc, n1, half);
      acc00 = wmma_bf16(a0, b0, acc00);
      acc01 = wmma_bf16(a0, b1, acc01);
      acc10 = wmma_bf16(a1, b0, acc10);
      acc11 = wmma_bf16(a1, b1, acc11);
    }
    __syncthreads();  // everyone done reading buf[ch&1] before overwriting it

    if (ch + 2 < 16) {
      unsigned ldsw = (ch & 1) ? lds1 : lds0;  // buffer (ch+2)&1 == ch&1
      unsigned vo = voff0 + 128u * (unsigned)(ch + 2);
      async_ld_b128(ldsw, vo, sbase);
      async_ld_b128(ldsw + 16, vo + 16, sbase);
    }
  }

  const float bm0 = bm[n0], bm1 = bm[n1];
#pragma unroll
  for (int r = 0; r < 8; ++r) {
    int rowA = mBase + mt0 * 16 + r + 8 * half;
    int rowB = rowA + 16;
    out[(size_t)rowA * DMM + n0] = acc00[r] + bm0;
    out[(size_t)rowA * DMM + n1] = acc01[r] + bm1;
    out[(size_t)rowB * DMM + n0] = acc10[r] + bm0;
    out[(size_t)rowB * DMM + n1] = acc11[r] + bm1;
  }
}

// ---------------------------------------------------------------------------
// Launch. Workspace layout (bf16):
//   w1b 3*128*256 | w2b 3*256*128 | Wmb 1024*1024 | hpk 16384*1024  (~34.4 MB)
// ---------------------------------------------------------------------------
extern "C" void kernel_launch(void* const* d_in, const int* in_sizes, int n_in,
                              void* d_out, int out_size, void* d_ws,
                              size_t ws_size, hipStream_t stream) {
  const float* x   = (const float*)d_in[0];
  const float* sg  = (const float*)d_in[1];
  const float* sb  = (const float*)d_in[2];
  const float* pg  = (const float*)d_in[3];
  const float* pb  = (const float*)d_in[4];
  const float* w1  = (const float*)d_in[5];
  const float* b1  = (const float*)d_in[6];
  const float* w2  = (const float*)d_in[7];
  const float* b2  = (const float*)d_in[8];
  const float* pkg = (const float*)d_in[9];
  const float* pkb = (const float*)d_in[10];
  const float* Wm  = (const float*)d_in[11];
  const float* bm  = (const float*)d_in[12];
  float* out = (float*)d_out;

  __bf16* w1b = (__bf16*)d_ws;
  __bf16* w2b = w1b + 3 * DD * HH;
  __bf16* wmb = w2b + 3 * HH * DD;
  __bf16* hpk = wmb + (size_t)DMM * DMM;

  swz_w1<<<(3 * DD * HH + 255) / 256, 256, 0, stream>>>(w1, w1b);
  swz_w2<<<(3 * HH * DD + 255) / 256, 256, 0, stream>>>(w2, w2b);
  swz_wm<<<(DMM * DMM) / 256, 256, 0, stream>>>(Wm, wmb);

  slotconv_encoder<<<BS_TOTAL / NS, 256, 0, stream>>>(
      x, sg, sb, pg, pb, w1b, b1, w2b, b2, pkg, pkb, hpk);

  slotconv_gemm<<<dim3(DMM / 128, BS_TOTAL / 64), 256, 0, stream>>>(hpk, wmb, bm,
                                                                    out);
}
